// GRU_GNN_Model_REFbf_52467320488318
// MI455X (gfx1250) — matrix operand order repaired and checked by reading, not compiled, not run on hardware
//
#include <hip/hip_runtime.h>
#include <hip/hip_bf16.h>
#include <math.h>

// ---------------------------------------------------------------------------
// GRU+GNN model on MI455X (gfx1250), wave32.
//  - All GEMMs via v_wmma_f32_16x16x32_bf16 (f32 accumulate)
//  - Weights / reused A-operands pre-packed into WMMA fragment layout (bf16)
//  - EdgeConv A-fragments packed in LDS at staging time (wide ds_load_b128)
//  - Edge indices copied global->LDS with GLOBAL_LOAD_ASYNC_TO_LDS (ASYNCcnt)
// ---------------------------------------------------------------------------

typedef __attribute__((ext_vector_type(16))) __bf16 v16bf;
typedef __attribute__((ext_vector_type(8)))  float  v8f;

#define B_SZ     16
#define N_VERTS  4000
#define N_EDGES  24000
#define IN_DIM   480
#define RFF_K    2160
#define RFF_OUT  4800
#define H1_DIM   256
#define OUT2     (N_VERTS * 16)   // 64000
#define GNN      128
#define FEAT_STRIDE 20            // 16 gru features + 3 pos + 1 pad

__device__ __forceinline__ float preluf(float x, float a) {
    return x >= 0.f ? x : a * x;
}

// segment-max via int atomics (valid with -inf init; sign-split trick)
__device__ __forceinline__ void atomic_max_f32(float* addr, float v) {
    if (v >= 0.f) atomicMax((int*)addr, __float_as_int(v));
    else          atomicMin((unsigned int*)addr, __float_as_uint(v));
}

__device__ __forceinline__ bool finitef(float v) {
    return (__float_as_uint(v) & 0x7F800000u) != 0x7F800000u;
}

// WMMA fragment index helpers (ISA 7.12.2, 16-bit operands, wave32):
//  A (16xK slab of 32): lane = m + 16*hi, elem e -> k = (e>>3)*16 + hi*8 + (e&7)
//  B (32x16):           lane = n + 16*hi, elem e -> k = (e>>3)*16 + hi*8 + (e&7)
//  C/D f32 16x16:       lane = n + 16*hi, vgpr v -> m = v + 8*hi

// ---------------------------------------------------------------------------
// K0: random Fourier features.  x_in = [cos(2pi x@B), sin(2pi x@B), x]
// ---------------------------------------------------------------------------
__global__ void rff_xin_kernel(const float* __restrict__ x,
                               const float* __restrict__ Brff,
                               float* __restrict__ xin) {
    int idx = blockIdx.x * blockDim.x + threadIdx.x;
    if (idx >= (RFF_K + IN_DIM) * B_SZ) return;
    int b   = idx & 15;
    int col = idx >> 4;
    if (col < RFF_K) {
        float s = 0.f;
        #pragma unroll 4
        for (int i = 0; i < IN_DIM; ++i)
            s += x[b * IN_DIM + i] * Brff[i * RFF_K + col];
        s *= 6.283185307179586f;
        xin[b * RFF_OUT + col]         = cosf(s);
        xin[b * RFF_OUT + RFF_K + col] = sinf(s);
    } else {
        int i = col - RFF_K;
        xin[b * RFF_OUT + 2 * RFF_K + i] = x[b * IN_DIM + i];
    }
}

// ---------------------------------------------------------------------------
// K1: h1 = prelu(x_in @ W1 + b1).  M=16(batch), N=256, K=4800.
// Tiny (16 waves total) -> direct gather loads are fine.
// ---------------------------------------------------------------------------
__global__ void mlp1_wmma_kernel(const float* __restrict__ xin,
                                 const float* __restrict__ W1,
                                 const float* __restrict__ b1,
                                 const float* __restrict__ a1,
                                 float* __restrict__ h1) {
    const int lane = threadIdx.x;
    const int n0   = blockIdx.x * 16;
    const int m    = lane & 15;
    const int hi   = lane >> 4;
    v8f acc = {};
    for (int k0 = 0; k0 < RFF_OUT; k0 += 32) {
        v16bf A, Bm;
        #pragma unroll
        for (int e = 0; e < 16; ++e) {
            int k = k0 + (e >> 3) * 16 + hi * 8 + (e & 7);
            A[e] = (__bf16)xin[m * RFF_OUT + k];
        }
        #pragma unroll
        for (int e = 0; e < 16; ++e) {
            int k = k0 + (e >> 3) * 16 + hi * 8 + (e & 7);
            Bm[e] = (__bf16)W1[k * H1_DIM + n0 + m];
        }
        acc = __builtin_amdgcn_wmma_f32_16x16x32_bf16(
            false, A, false, Bm, (short)0, acc, false, false);
    }
    const int n = n0 + m;
    const float bn = b1[n], an = a1[n];
    #pragma unroll
    for (int v = 0; v < 8; ++v) {
        int b = v + 8 * hi;
        h1[b * H1_DIM + n] = preluf(acc[v] + bn, an);
    }
}

// ---------------------------------------------------------------------------
// K2: pre-pack a [K_TOT x 128] weight matrix into bf16 WMMA B-fragments.
// out fragment layout: [(ks*8 + nt)*32 + lane] -> v16bf (K padded with 0).
// ---------------------------------------------------------------------------
__global__ void pack_bfrag_kernel(const float* __restrict__ W,
                                  __bf16* __restrict__ out,
                                  int K_TOT, int ksteps) {
    int idx = blockIdx.x * blockDim.x + threadIdx.x;   // frag*32 + lane
    if (idx >= ksteps * 8 * 32) return;
    int lane = idx & 31;
    int frag = idx >> 5;
    int nt = frag & 7, ks = frag >> 3;
    int m = lane & 15, hi = lane >> 4;
    __bf16* dst = out + idx * 16;
    #pragma unroll
    for (int e = 0; e < 16; ++e) {
        int k = ks * 32 + (e >> 3) * 16 + hi * 8 + (e & 7);
        float v = (k < K_TOT) ? W[k * GNN + nt * 16 + m] : 0.f;
        dst[e] = (__bf16)v;
    }
}

// ---------------------------------------------------------------------------
// K3: pre-pack h1 [16 x 256] into bf16 WMMA A-fragments (shared by all
// 4000 MLP2 tiles).  out: [ks*32 + lane] -> v16bf.
// ---------------------------------------------------------------------------
__global__ void pack_h1_kernel(const float* __restrict__ h1,
                               __bf16* __restrict__ out) {
    int idx = blockIdx.x * blockDim.x + threadIdx.x;   // ks*32 + lane, 256 total
    if (idx >= (H1_DIM / 32) * 32) return;
    int lane = idx & 31, ks = idx >> 5;
    int m = lane & 15, hi = lane >> 4;
    __bf16* dst = out + idx * 16;
    #pragma unroll
    for (int e = 0; e < 16; ++e) {
        int k = ks * 32 + (e >> 3) * 16 + hi * 8 + (e & 7);
        dst[e] = (__bf16)h1[m * H1_DIM + k];
    }
}

// ---------------------------------------------------------------------------
// K4: write vertex positions into feature buffer [B][N][20], zero the pad.
// ---------------------------------------------------------------------------
__global__ void feat_pos_kernel(const float* __restrict__ pos,
                                float* __restrict__ feat) {
    int idx = blockIdx.x * blockDim.x + threadIdx.x;
    if (idx >= B_SZ * N_VERTS) return;
    feat[idx * FEAT_STRIDE + 16] = pos[idx * 3 + 0];
    feat[idx * FEAT_STRIDE + 17] = pos[idx * 3 + 1];
    feat[idx * FEAT_STRIDE + 18] = pos[idx * 3 + 2];
    feat[idx * FEAT_STRIDE + 19] = 0.f;
}

// ---------------------------------------------------------------------------
// K5: gru_out = prelu(h1 @ W2 + b2) into feat[b][vert][0..15].
// M=16(batch), N=64000, K=256.  One 16-col tile == one vertex.
// A from pre-packed fragments; W2 (65.5 MB) streamed once -> HBM roofline.
// ---------------------------------------------------------------------------
__global__ void mlp2_wmma_kernel(const v16bf* __restrict__ Ap,
                                 const float* __restrict__ W2,
                                 const float* __restrict__ b2,
                                 const float* __restrict__ a2,
                                 float* __restrict__ feat) {
    const int lane = threadIdx.x & 31;
    const int wave = threadIdx.x >> 5;
    const int tile = blockIdx.x * 8 + wave;      // vertex index
    const int m    = lane & 15;
    const int hi   = lane >> 4;
    const int n0   = tile * 16;
    v8f acc = {};
    #pragma unroll
    for (int k0 = 0; k0 < H1_DIM; k0 += 32) {
        if (k0 + 32 < H1_DIM)   // pull next K-slab of streamed W2 (global_prefetch_b8)
            __builtin_prefetch(&W2[(size_t)(k0 + 32 + hi * 16) * OUT2 + n0 + m], 0, 1);
        v16bf A = Ap[(k0 >> 5) * 32 + lane];
        v16bf Bm;
        #pragma unroll
        for (int e = 0; e < 16; ++e) {
            int k = k0 + (e >> 3) * 16 + hi * 8 + (e & 7);
            Bm[e] = (__bf16)W2[(size_t)k * OUT2 + n0 + m];
        }
        acc = __builtin_amdgcn_wmma_f32_16x16x32_bf16(
            false, A, false, Bm, (short)0, acc, false, false);
    }
    const int n = n0 + m;
    const float bn = b2[n], an = a2[n];
    #pragma unroll
    for (int v = 0; v < 8; ++v) {
        int b = v + 8 * hi;
        feat[(size_t)(b * N_VERTS + tile) * FEAT_STRIDE + m] = preluf(acc[v] + bn, an);
    }
}

// ---------------------------------------------------------------------------
// K6: fill f32 buffer with -inf (segment-max identity), float4 stores.
// ---------------------------------------------------------------------------
__global__ void fill_neginf_kernel(float4* __restrict__ p, int n4) {
    int idx = blockIdx.x * blockDim.x + threadIdx.x;
    if (idx < n4) {
        float ninf = -__builtin_inff();
        p[idx] = make_float4(ninf, ninf, ninf, ninf);
    }
}

// ---------------------------------------------------------------------------
// K7: EdgeConv.  One wave = 16 edges x 128 output cols.
//   A row (edge e): [ x_dst (F_IN) , x_src - x_dst (F_IN) ], K padded to 32.
//   Edge indices:  GLOBAL_LOAD_ASYNC_TO_LDS + s_wait_asynccnt.
//   A fragments:   packed bf16 in LDS at staging time -> ds_load_b128 pairs.
//   B fragments:   pre-packed bf16 in global (L2 resident) -> b128 pairs.
//   segment_max:   f32 atomic-max into -inf-initialized agg.
// ---------------------------------------------------------------------------
template <int F_IN, int F_STRIDE>
__global__ void edgeconv_wmma_kernel(const float* __restrict__ feat,
                                     const int* __restrict__ srcIdx,
                                     const int* __restrict__ dstIdx,
                                     const v16bf* __restrict__ Wp,     // packed B frags
                                     const float* __restrict__ bias,
                                     const float* __restrict__ alpha,
                                     float* __restrict__ agg) {        // [B][N][128]
    constexpr int K_TOT  = 2 * F_IN;
    constexpr int KSTEPS = (K_TOT + 31) / 32;
    __shared__ __align__(32) __bf16 sA[KSTEPS * 32 * 16];
    __shared__ int sdst[16];
    __shared__ int ssrc[16];

    const int lane = threadIdx.x;      // 0..31
    const int tile = blockIdx.x;       // 0..1499 (24000/16 exact)
    const int b    = blockIdx.y;       // 0..15

    // --- async global->LDS copy of the 16 dst + 16 src indices -------------
    {
        const int* gp = (lane < 16) ? (dstIdx + tile * 16 + (lane & 15))
                                    : (srcIdx + tile * 16 + (lane & 15));
        unsigned ldsoff = (unsigned)(size_t)((lane < 16) ? (void*)&sdst[lane & 15]
                                                         : (void*)&ssrc[lane & 15]);
        asm volatile("global_load_async_to_lds_b32 %0, %1, off"
                     :: "v"(ldsoff), "v"(gp) : "memory");
        asm volatile("s_wait_asynccnt 0x0" ::: "memory");
    }
    __syncthreads();

    // --- zero the packed-A pad region (only when K_TOT % 32 != 0) ----------
    if constexpr ((K_TOT & 31) != 0) {
        unsigned* z = (unsigned*)sA;
        for (int i = lane; i < (KSTEPS * 32 * 16) / 2; i += 32) z[i] = 0u;
        __syncthreads();
    }

    // --- gather edge features, pack straight into bf16 A-fragment layout ---
    auto packStore = [&](int eEdge, int kk, float val) {
        int ks = kk >> 5, kl = kk & 31;
        int hi2 = (kl >> 3) & 1;
        int e   = (kl >> 4) * 8 + (kl & 7);
        sA[((ks * 32) + eEdge + 16 * hi2) * 16 + e] = (__bf16)val;
    };
    if constexpr ((F_IN & 3) == 0) {
        for (int i = lane; i < 16 * (F_IN / 4); i += 32) {
            int e  = i / (F_IN / 4);
            int k4 = (i - e * (F_IN / 4)) * 4;
            const float4 xi = *(const float4*)&feat[(size_t)(b * N_VERTS + sdst[e]) * F_STRIDE + k4];
            const float4 xj = *(const float4*)&feat[(size_t)(b * N_VERTS + ssrc[e]) * F_STRIDE + k4];
            packStore(e, k4 + 0, xi.x); packStore(e, k4 + 0 + F_IN, xj.x - xi.x);
            packStore(e, k4 + 1, xi.y); packStore(e, k4 + 1 + F_IN, xj.y - xi.y);
            packStore(e, k4 + 2, xi.z); packStore(e, k4 + 2 + F_IN, xj.z - xi.z);
            packStore(e, k4 + 3, xi.w); packStore(e, k4 + 3 + F_IN, xj.w - xi.w);
        }
    } else {
        for (int i = lane; i < 16 * F_IN; i += 32) {
            int e = i / F_IN, k = i - e * F_IN;
            float xi = feat[(size_t)(b * N_VERTS + sdst[e]) * F_STRIDE + k];
            float xj = feat[(size_t)(b * N_VERTS + ssrc[e]) * F_STRIDE + k];
            packStore(e, k, xi);
            packStore(e, k + F_IN, xj - xi);
        }
    }
    __syncthreads();

    // --- WMMA main loop: 1 A-frag ds load + 8 B-frag b128 loads per k-step --
    const int m  = lane & 15;
    const int hi = lane >> 4;
    const v16bf* sAv = (const v16bf*)sA;
    v8f acc[8] = {};
    #pragma unroll
    for (int ks = 0; ks < KSTEPS; ++ks) {
        v16bf A = sAv[ks * 32 + lane];
        #pragma unroll
        for (int nt = 0; nt < 8; ++nt) {
            v16bf Bm = Wp[(ks * 8 + nt) * 32 + lane];
            acc[nt] = __builtin_amdgcn_wmma_f32_16x16x32_bf16(
                false, A, false, Bm, (short)0, acc[nt], false, false);
        }
    }

    // --- bias + PReLU + segment-max scatter --------------------------------
    #pragma unroll
    for (int nt = 0; nt < 8; ++nt) {
        int n = nt * 16 + m;
        float bn = bias[n], an = alpha[n];
        #pragma unroll
        for (int v = 0; v < 8; ++v) {
            int e = v + 8 * hi;               // C/D layout: edge row = v + 8*hi
            float val = preluf(acc[nt][v] + bn, an);
            atomic_max_f32(&agg[(size_t)(b * N_VERTS + sdst[e]) * GNN + n], val);
        }
    }
}

// ---------------------------------------------------------------------------
// K8: replace non-finite (-inf from empty segments) with 0 in place.
// ---------------------------------------------------------------------------
__global__ void finalize_kernel(float* __restrict__ p, int n) {
    int idx = blockIdx.x * blockDim.x + threadIdx.x;
    if (idx < n) {
        float v = p[idx];
        p[idx] = finitef(v) ? v : 0.f;
    }
}

// ---------------------------------------------------------------------------
// K9: y = (finite(x1) ? x1 : 0) @ Wo + bo    -> [B, N*3]
// ---------------------------------------------------------------------------
__global__ void out_kernel(const float* __restrict__ x1,
                           const float* __restrict__ Wo,
                           const float* __restrict__ bo,
                           float* __restrict__ y) {
    int idx = blockIdx.x * blockDim.x + threadIdx.x;  // b*N_VERTS + v
    if (idx >= B_SZ * N_VERTS) return;
    float s0 = bo[0], s1 = bo[1], s2 = bo[2];
    const float* row = &x1[(size_t)idx * GNN];
    #pragma unroll 4
    for (int c = 0; c < GNN; ++c) {
        float xv = row[c];
        if (!finitef(xv)) xv = 0.f;
        s0 += xv * Wo[c * 3 + 0];
        s1 += xv * Wo[c * 3 + 1];
        s2 += xv * Wo[c * 3 + 2];
    }
    y[idx * 3 + 0] = s0;
    y[idx * 3 + 1] = s1;
    y[idx * 3 + 2] = s2;
}

// ---------------------------------------------------------------------------
extern "C" void kernel_launch(void* const* d_in, const int* in_sizes, int n_in,
                              void* d_out, int out_size, void* d_ws, size_t ws_size,
                              hipStream_t stream) {
    const float* x    = (const float*)d_in[0];
    const float* pos  = (const float*)d_in[1];
    const int*   eidx = (const int*)  d_in[2];
    const float* Brff = (const float*)d_in[3];
    const float* W1   = (const float*)d_in[4];
    const float* b1   = (const float*)d_in[5];
    const float* a1   = (const float*)d_in[6];
    const float* W2   = (const float*)d_in[7];
    const float* b2   = (const float*)d_in[8];
    const float* a2   = (const float*)d_in[9];
    const float* We1  = (const float*)d_in[10];
    const float* be1  = (const float*)d_in[11];
    const float* ae1  = (const float*)d_in[12];
    const float* We2  = (const float*)d_in[13];
    const float* be2  = (const float*)d_in[14];
    const float* ae2  = (const float*)d_in[15];
    const float* Wo   = (const float*)d_in[16];
    const float* bo   = (const float*)d_in[17];
    float* out = (float*)d_out;

    // workspace carve-up (floats; base is 256B-aligned, all offsets 32B-aligned):
    float* ws   = (float*)d_ws;
    float* xin  = ws;                       // 16*4800      =    76,800
    float* h1   = xin  + 76800;             // 16*256       =     4,096
    float* feat = h1   + 4096;              // 16*4000*20   = 1,280,000
    float* x0   = feat + 1280000;           // 16*4000*128  = 8,192,000
    float* x1   = x0   + 8192000;           // 16*4000*128  = 8,192,000
    __bf16* wp1 = (__bf16*)(x1 + 8192000);  // 2*8*32*16    =     8,192 bf16
    __bf16* wp2 = wp1 + 2 * 8 * 32 * 16;    // 8*8*32*16    =    32,768 bf16
    __bf16* h1p = wp2 + 8 * 8 * 32 * 16;    // 8*32*16      =     4,096 bf16

    const int* srcIdx = eidx;               // edge_index[0]
    const int* dstIdx = eidx + N_EDGES;     // edge_index[1]

    const int AGG_ELEMS = B_SZ * N_VERTS * GNN;   // 8,192,000

    rff_xin_kernel<<<((RFF_K + IN_DIM) * B_SZ + 255) / 256, 256, 0, stream>>>(x, Brff, xin);
    mlp1_wmma_kernel<<<H1_DIM / 16, 32, 0, stream>>>(xin, W1, b1, a1, h1);
    pack_h1_kernel<<<1, 256, 0, stream>>>(h1, h1p);
    pack_bfrag_kernel<<<2, 256, 0, stream>>>(We1, wp1, 2 * 19, 2);
    pack_bfrag_kernel<<<8, 256, 0, stream>>>(We2, wp2, 2 * 128, 8);
    feat_pos_kernel<<<(B_SZ * N_VERTS + 255) / 256, 256, 0, stream>>>(pos, feat);
    mlp2_wmma_kernel<<<N_VERTS / 8, 256, 0, stream>>>((const v16bf*)h1p, W2, b2, a2, feat);

    fill_neginf_kernel<<<(AGG_ELEMS / 4 + 255) / 256, 256, 0, stream>>>((float4*)x0, AGG_ELEMS / 4);
    edgeconv_wmma_kernel<19, FEAT_STRIDE>
        <<<dim3(N_EDGES / 16, B_SZ), 32, 0, stream>>>(feat, srcIdx, dstIdx,
                                                      (const v16bf*)wp1, be1, ae1, x0);
    finalize_kernel<<<(AGG_ELEMS + 255) / 256, 256, 0, stream>>>(x0, AGG_ELEMS);

    fill_neginf_kernel<<<(AGG_ELEMS / 4 + 255) / 256, 256, 0, stream>>>((float4*)x1, AGG_ELEMS / 4);
    edgeconv_wmma_kernel<128, 128>
        <<<dim3(N_EDGES / 16, B_SZ), 32, 0, stream>>>(x0, srcIdx, dstIdx,
                                                      (const v16bf*)wp2, be2, ae2, x1);

    out_kernel<<<(B_SZ * N_VERTS + 255) / 256, 256, 0, stream>>>(x1, Wo, bo, out);
}
